// Model_17506286698644
// MI455X (gfx1250) — matrix operand order repaired and checked by reading, not compiled, not run on hardware
//
#include <hip/hip_runtime.h>
#include <math.h>

// ---------- WMMA types / helpers (gfx1250, wave32) ----------
typedef __attribute__((ext_vector_type(16))) __bf16 v16bf;
typedef __attribute__((ext_vector_type(8)))  float  v8f;
typedef __attribute__((ext_vector_type(4)))  unsigned int u32x4;

union FragU {
    v16bf v;
    u32x4 q[2];
    unsigned short u[16];
};

static __device__ __forceinline__ unsigned short f32_to_bf16(float f) {
    unsigned u = __float_as_uint(f);
    unsigned r = u + 0x7FFFu + ((u >> 16) & 1u);   // round-to-nearest-even
    return (unsigned short)(r >> 16);
}

static __device__ __forceinline__ float sigf(float x) {
    return 1.0f / (1.0f + __expf(-x));
}

// A fragment: 16x32 (MxK) bf16, source row-major [row][ldk], rows m0..m0+15.
// Layout: lanes 0-15 hold M=lane, K=k0..k0+15;
//         lanes 16-31 hold M=lane-16, K=k0+16..k0+31.
static __device__ __forceinline__ v16bf load_a_frag(const unsigned short* base, int ldk,
                                                    int m0, int k0, int lane) {
    int r = lane & 15;
    int g = lane >> 4;
    const unsigned short* p = base + (size_t)(m0 + r) * ldk + k0 + g * 16;
    FragU f;
    f.q[0] = *(const u32x4*)(p);        // 8 bf16: K = k0+g*16 .. +7
    f.q[1] = *(const u32x4*)(p + 8);    // 8 bf16: K = k0+g*16+8 .. +15
    return f.v;
}

// B fragment (unguarded): 32x16 (KxN) bf16 from weights W row-major [n][ldk]
// (B[k][n] = W[n][k]). Lane holds column n0+(lane&15), K run g*16..g*16+15.
static __device__ __forceinline__ v16bf load_b_nog(const unsigned short* W, int ldk,
                                                   int n0, int k0, int lane) {
    int n = n0 + (lane & 15);
    int g = lane >> 4;
    const unsigned short* p = W + (size_t)n * ldk + k0 + g * 16;
    FragU f;
    f.q[0] = *(const u32x4*)(p);
    f.q[1] = *(const u32x4*)(p + 8);
    return f.v;
}

// B fragment with row guard (only for the ragged-N logits GEMM).
static __device__ __forceinline__ v16bf load_b_grd(const unsigned short* W, int ldk,
                                                   int n0, int k0, int lane, int nreal) {
    int n = n0 + (lane & 15);
    int g = lane >> 4;
    FragU f;
    if (n < nreal) {
        const unsigned short* p = W + (size_t)n * ldk + k0 + g * 16;
        f.q[0] = *(const u32x4*)(p);
        f.q[1] = *(const u32x4*)(p + 8);
    } else {
        u32x4 z = {0u, 0u, 0u, 0u};
        f.q[0] = z;
        f.q[1] = z;
    }
    return f.v;
}

static __device__ __forceinline__ v8f wmma_bf16(v16bf a, v16bf b, v8f c) {
    return __builtin_amdgcn_wmma_f32_16x16x32_bf16(false, a, false, b, (short)0, c,
                                                   false, false);
}

// Defeat cross-timestep hoisting of loop-invariant weight loads WITHOUT
// destroying address-space inference: launder an integer offset (stays 0),
// keeping the pointer chain rooted at the kernarg (=> global_load, not flat).
#define OPAQUE_OFF(x) asm volatile("" : "+s"(x))

// ---------- small utility kernels ----------
__global__ void k_cvt_bf16(const float* __restrict__ in, unsigned short* __restrict__ out,
                           int n) {
    int i = blockIdx.x * 256 + threadIdx.x;
    if (i < n) out[i] = f32_to_bf16(in[i]);
}

// Fold time-pairs into feature dim and convert to bf16.
// out X[(t*64 + b)*2F + j]; j<F from src(b, 2t, j) else src(b, 2t+1, j-F).
// src element (b, t2, f) at address b*sb + t2*st + f.
__global__ void k_fold_bf16(const float* __restrict__ src, unsigned short* __restrict__ X,
                            int F, long sb, long st, int total) {
    int i = blockIdx.x * 256 + threadIdx.x;
    if (i >= total) return;
    int F2 = 2 * F;
    int j  = i % F2;
    int rb = i / F2;           // t*64 + b
    int b  = rb & 63;
    int t  = rb >> 6;
    int t2 = 2 * t + (j >= F ? 1 : 0);
    int f  = (j >= F) ? (j - F) : j;
    X[i] = f32_to_bf16(src[(size_t)b * sb + (size_t)t2 * st + f]);
}

// Xemb[(s*64+b)*256 + h] = bf16(emb[seqs[b][s]][h]);  seqs: [64][128] int32
__global__ void k_gather_emb(const float* __restrict__ emb, const int* __restrict__ seqs,
                             unsigned short* __restrict__ X) {
    int i = blockIdx.x * 256 + threadIdx.x;
    if (i >= 128 * 64 * 256) return;
    int h  = i & 255;
    int rb = i >> 8;           // s*64 + b
    int b  = rb & 63;
    int s  = rb >> 6;
    int tok = seqs[b * 128 + s];
    X[i] = f32_to_bf16(emb[(size_t)tok * 256 + h]);
}

// ---------- generic bf16 WMMA GEMM:  C[m][n] = sum_k A[m][k] * W[n][k] + bias ----------
// Weights-stationary: each wave computes 4 consecutive M-tiles of one N-tile,
// so one B-fragment load feeds 4 WMMAs. All 4 A-fragments are loaded before the
// WMMA chain so the loads clause together. M must be a multiple of 64.
template <bool GUARD>
__global__ void __launch_bounds__(256) k_gemm(
    const unsigned short* __restrict__ A, int lda,
    const unsigned short* __restrict__ W, int ldw,
    const float* __restrict__ bias0, const float* __restrict__ bias1,
    float* __restrict__ C, int ldc,
    int M, int Nreal, int Npad, int K) {
    int wave = threadIdx.x >> 5;
    int lane = threadIdx.x & 31;
    int ntn  = Npad >> 4;
    int tile = blockIdx.x * 8 + wave;
    int total = (M >> 6) * ntn;
    if (tile >= total) return;
    int mg = tile / ntn, nt = tile - mg * ntn;
    int m0 = mg << 6, n0 = nt << 4;
    v8f acc[4];
#pragma unroll
    for (int i = 0; i < 4; ++i) acc[i] = v8f{};
    for (int k0 = 0; k0 < K; k0 += 32) {
        v16bf b = GUARD ? load_b_grd(W, ldw, n0, k0, lane, Nreal)
                        : load_b_nog(W, ldw, n0, k0, lane);
        v16bf aa[4];
#pragma unroll
        for (int i = 0; i < 4; ++i)
            aa[i] = load_a_frag(A, lda, m0 + i * 16, k0, lane);
#pragma unroll
        for (int i = 0; i < 4; ++i)
            acc[i] = wmma_bf16(aa[i], b, acc[i]);
    }
    int col = n0 + (lane & 15);
    if (!GUARD || col < Nreal) {
        float bv = (bias0 ? bias0[col] : 0.0f) + (bias1 ? bias1[col] : 0.0f);
#pragma unroll
        for (int i = 0; i < 4; ++i) {
            int rbase = m0 + i * 16 + ((lane >> 4) << 3);
#pragma unroll
            for (int v = 0; v < 8; ++v)
                C[(size_t)(rbase + v) * ldc + col] = acc[i][v] + bv;
        }
    }
}

// ---------- persistent BiLSTM recurrence ----------
// grid = 8 blocks: blockIdx = dir*4 + batch_group. Each block owns 16 batch rows,
// loops over all timesteps; per step: g = G_pre[t] + h@Whh^T (WMMA, h in LDS bf16),
// K outer / N-tiles inner so one LDS A-fragment feeds NT WMMAs.
// Whh is staged in LDS (fully for H=64/128; first NSROWS gate rows for H=256);
// the unstaged remainder streams from L2 behind an anti-hoist offset barrier.
template <int H>
__global__ void __launch_bounds__(512) k_lstm(
    const float* __restrict__ Gf, const float* __restrict__ Gb,
    const unsigned short* __restrict__ WhhF, const unsigned short* __restrict__ WhhB,
    float* __restrict__ Y, int Tsteps) {
    constexpr int N      = 4 * H;
    constexpr int NT     = H / 64;                 // N-tiles per wave
    constexpr int NSROWS = (H == 256) ? 384 : N;   // weight rows staged in LDS
    __shared__ __attribute__((aligned(16))) float gbuf[16 * N];
    __shared__ __attribute__((aligned(16))) unsigned short hbuf[16 * H];
    __shared__ __attribute__((aligned(16))) unsigned short wlds[NSROWS * H];
    int dir = blockIdx.x >> 2;
    int bg  = blockIdx.x & 3;
    const float* G = dir ? Gb : Gf;
    const unsigned short* Whh = dir ? WhhB : WhhF;
    int tid = threadIdx.x, wave = tid >> 5, lane = tid & 31;
    float creg[8];
#pragma unroll
    for (int i = 0; i < 8; ++i) creg[i] = 0.0f;
    for (int i = tid; i < 16 * H; i += 512) hbuf[i] = 0;
    // stage weights into LDS (16B vector copies)
    for (int i = tid; i < (NSROWS * H) / 8; i += 512)
        ((u32x4*)wlds)[i] = ((const u32x4*)Whh)[i];
    __syncthreads();
    for (int step = 0; step < Tsteps; ++step) {
        int t = dir ? (Tsteps - 1 - step) : step;
        // prefetch next step's precomputed gates (global_prefetch_b8)
        {
            int stepn = (step + 1 < Tsteps) ? step + 1 : step;
            int tn = dir ? (Tsteps - 1 - stepn) : stepn;
            const char* pf = (const char*)(G + ((size_t)tn * 64 + bg * 16) * N);
            for (int off = tid * 128; off < 16 * N * 4; off += 512 * 128)
                __builtin_prefetch(pf + off, 0, 1);
        }
        size_t woff = 0;
        OPAQUE_OFF(woff);
        const unsigned short* Wg = Whh + woff;
        // gates GEMM: [16,H] @ [H,4H]
        v8f acc[NT];
#pragma unroll
        for (int i = 0; i < NT; ++i) acc[i] = v8f{};
#pragma unroll
        for (int k0 = 0; k0 < H; k0 += 32) {
            v16bf a = load_a_frag(hbuf, H, 0, k0, lane);
            v16bf bb[NT];
#pragma unroll
            for (int i = 0; i < NT; ++i) {
                int n0 = (wave + i * 16) << 4;
                bb[i] = (n0 + 16 <= NSROWS)
                            ? load_b_nog(wlds, H, n0, k0, lane)
                            : load_b_nog(Wg, H, n0, k0, lane);
            }
#pragma unroll
            for (int i = 0; i < NT; ++i)
                acc[i] = wmma_bf16(a, bb[i], acc[i]);
        }
#pragma unroll
        for (int i = 0; i < NT; ++i) {
            int col = ((wave + i * 16) << 4) + (lane & 15);
            int rbase = (lane >> 4) << 3;
#pragma unroll
            for (int v = 0; v < 8; ++v) {
                int r = rbase + v;
                gbuf[r * N + col] =
                    acc[i][v] + G[((size_t)t * 64 + bg * 16 + r) * N + col];
            }
        }
        __syncthreads();
        // LSTM pointwise (gate order i,f,g,o)
        {
            int ci = 0;
            for (int idx = tid; idx < 16 * H; idx += 512, ++ci) {
                int b = idx / H, j = idx - b * H;
                float gi  = gbuf[b * N + j];
                float gfv = gbuf[b * N + H + j];
                float gg  = gbuf[b * N + 2 * H + j];
                float go  = gbuf[b * N + 3 * H + j];
                float c = sigf(gfv) * creg[ci] + sigf(gi) * tanhf(gg);
                creg[ci] = c;
                float h = sigf(go) * tanhf(c);
                hbuf[b * H + j] = f32_to_bf16(h);
                Y[((size_t)t * 64 + bg * 16 + b) * (2 * H) + dir * H + j] = h;
            }
        }
        __syncthreads();
    }
}

// ---------- persistent attention decoder ----------
// grid = 4 blocks, each owns 16 batch rows; loops s=0..127.
// Per step: g = Gx[s] + ctx@WihC^T + h@Whh^T; LSTM; q = h@Wq^T + bq;
// scores = keys.q; softmax; ctx = values.w; log [h,ctx] (bf16) for final logits GEMM.
// Wq staged in LDS; Wih/Whh stream from L2 behind anti-hoist offset barriers.
__global__ void __launch_bounds__(512) k_dec(
    const float* __restrict__ Gx, const float* __restrict__ keys,
    const float* __restrict__ vals,
    const unsigned short* __restrict__ Wih,   // [1024][512] bf16 (ctx part = cols 256..511)
    const unsigned short* __restrict__ Whh,   // [1024][256] bf16
    const unsigned short* __restrict__ Wq,    // [256][256]  bf16
    const float* __restrict__ bq,
    unsigned short* __restrict__ hc) {        // [64][128][512] bf16
    __shared__ __attribute__((aligned(16))) float gbuf[16 * 1024];
    __shared__ __attribute__((aligned(16))) unsigned short hbuf[16 * 256];
    __shared__ __attribute__((aligned(16))) unsigned short ctxbuf[16 * 256];
    __shared__ __attribute__((aligned(16))) unsigned short wqlds[256 * 256];
    float* qbuf = gbuf;             // reuse (gbuf fully consumed before q phase)
    float* sbuf = gbuf + 16 * 256;  // reuse
    int bg = blockIdx.x;
    int tid = threadIdx.x, wave = tid >> 5, lane = tid & 31;
    float creg[8];
#pragma unroll
    for (int i = 0; i < 8; ++i) creg[i] = 0.0f;
    for (int i = tid; i < 16 * 256; i += 512) { hbuf[i] = 0; ctxbuf[i] = 0; }
    for (int i = tid; i < (256 * 256) / 8; i += 512)
        ((u32x4*)wqlds)[i] = ((const u32x4*)Wq)[i];
    __syncthreads();
    for (int s = 0; s < 128; ++s) {
        // prefetch next step's precomputed gates
        {
            int sn = (s + 1 < 128) ? s + 1 : s;
            const char* pf = (const char*)(Gx + ((size_t)sn * 64 + bg * 16) * 1024);
            for (int off = tid * 128; off < 16 * 1024 * 4; off += 512 * 128)
                __builtin_prefetch(pf + off, 0, 1);
        }
        size_t woff = 0;
        OPAQUE_OFF(woff);
        const unsigned short* WihG = Wih + 256 + woff;
        const unsigned short* WhhG = Whh + woff;
        // gates: ctx part + h part, N = 1024 (4 N-tiles per wave, K outer)
        v8f acc[4];
#pragma unroll
        for (int i = 0; i < 4; ++i) acc[i] = v8f{};
#pragma unroll
        for (int k0 = 0; k0 < 256; k0 += 32) {
            v16bf a = load_a_frag(ctxbuf, 256, 0, k0, lane);
            v16bf bb[4];
#pragma unroll
            for (int i = 0; i < 4; ++i)
                bb[i] = load_b_nog(WihG, 512, (wave + i * 16) << 4, k0, lane);
#pragma unroll
            for (int i = 0; i < 4; ++i)
                acc[i] = wmma_bf16(a, bb[i], acc[i]);
        }
#pragma unroll
        for (int k0 = 0; k0 < 256; k0 += 32) {
            v16bf a = load_a_frag(hbuf, 256, 0, k0, lane);
            v16bf bb[4];
#pragma unroll
            for (int i = 0; i < 4; ++i)
                bb[i] = load_b_nog(WhhG, 256, (wave + i * 16) << 4, k0, lane);
#pragma unroll
            for (int i = 0; i < 4; ++i)
                acc[i] = wmma_bf16(a, bb[i], acc[i]);
        }
#pragma unroll
        for (int i = 0; i < 4; ++i) {
            int col = ((wave + i * 16) << 4) + (lane & 15);
            int rbase = (lane >> 4) << 3;
#pragma unroll
            for (int v = 0; v < 8; ++v) {
                int r = rbase + v;
                gbuf[r * 1024 + col] =
                    acc[i][v] + Gx[((size_t)s * 64 + bg * 16 + r) * 1024 + col];
            }
        }
        __syncthreads();
        // LSTM pointwise, H = 256
        {
            int ci = 0;
            for (int idx = tid; idx < 16 * 256; idx += 512, ++ci) {
                int b = idx >> 8, j = idx & 255;
                float gi  = gbuf[b * 1024 + j];
                float gfv = gbuf[b * 1024 + 256 + j];
                float gg  = gbuf[b * 1024 + 512 + j];
                float go  = gbuf[b * 1024 + 768 + j];
                float c = sigf(gfv) * creg[ci] + sigf(gi) * tanhf(gg);
                creg[ci] = c;
                float h = sigf(go) * tanhf(c);
                unsigned short hb = f32_to_bf16(h);
                hbuf[b * 256 + j] = hb;
                hc[(((size_t)(bg * 16 + b)) * 128 + s) * 512 + j] = hb;
            }
        }
        __syncthreads();
        // q = h@Wq^T + bq (16 tiles, one per wave; Wq in LDS)
        {
            int n0 = wave << 4;
            v8f qa = {};
#pragma unroll
            for (int k0 = 0; k0 < 256; k0 += 32) {
                v16bf a = load_a_frag(hbuf, 256, 0, k0, lane);
                v16bf b = load_b_nog(wqlds, 256, n0, k0, lane);
                qa = wmma_bf16(a, b, qa);
            }
            int col = n0 + (lane & 15);
            int rbase = (lane >> 4) << 3;
#pragma unroll
            for (int v = 0; v < 8; ++v)
                qbuf[(rbase + v) * 256 + col] = qa[v] + bq[col];
        }
        __syncthreads();
        // scores[b][t] = keys[t][bglob] . q[b]
        for (int idx = tid; idx < 16 * 256; idx += 512) {
            int b = idx >> 8, t = idx & 255;
            const float* kp = keys + ((size_t)t * 64 + bg * 16 + b) * 256;
            const float* qp = qbuf + b * 256;
            float a = 0.0f;
            for (int h = 0; h < 256; ++h) a += kp[h] * qp[h];
            sbuf[idx] = a;
        }
        __syncthreads();
        // softmax over t (one row per wave)
        {
            int b = wave;
            float m = -3.4e38f;
            for (int t = lane; t < 256; t += 32) m = fmaxf(m, sbuf[b * 256 + t]);
            for (int off = 16; off > 0; off >>= 1) m = fmaxf(m, __shfl_xor(m, off, 32));
            float sum = 0.0f;
            for (int t = lane; t < 256; t += 32) {
                float e = __expf(sbuf[b * 256 + t] - m);
                sbuf[b * 256 + t] = e;
                sum += e;
            }
            for (int off = 16; off > 0; off >>= 1) sum += __shfl_xor(sum, off, 32);
            float inv = 1.0f / sum;
            for (int t = lane; t < 256; t += 32) sbuf[b * 256 + t] *= inv;
        }
        __syncthreads();
        // ctx[b][h] = sum_t vals[t][bglob][h] * w[b][t]
        for (int idx = tid; idx < 16 * 256; idx += 512) {
            int b = idx >> 8, h = idx & 255;
            const float* wp = sbuf + b * 256;
            const float* vp = vals + ((size_t)(bg * 16 + b)) * 256 + h;
            float a = 0.0f;
            for (int t = 0; t < 256; ++t) a += vp[(size_t)t * 16384] * wp[t];
            unsigned short cb = f32_to_bf16(a);
            ctxbuf[b * 256 + h] = cb;
            hc[(((size_t)(bg * 16 + b)) * 128 + s) * 512 + 256 + h] = cb;
        }
        __syncthreads();
    }
}

// ---------- host orchestration ----------
extern "C" void kernel_launch(void* const* d_in, const int* in_sizes, int n_in,
                              void* d_out, int out_size, void* d_ws, size_t ws_size,
                              hipStream_t stream) {
    if (n_in < 39) return;
    const size_t WS_NEEDED = 264241152ull;
    if (ws_size < WS_NEEDED) return;

    const float* spectras = (const float*)d_in[0];
    const int*   seqs     = (const int*)d_in[1];

    char* ws = (char*)d_ws;
    float*          Gf    = (float*)(ws);                       // 67,108,864 B
    float*          Gb    = (float*)(ws + 67108864);            // 67,108,864 B
    unsigned short* xbuf  = (unsigned short*)(ws + 134217728);  // 20,971,520 B
    float*          ybufA = (float*)(ws + 155189248);           // 33,554,432 B
    float*          ybufB = (float*)(ws + 188743680);           // 33,554,432 B
    float*          keys  = (float*)(ws + 222298112);           // 16,777,216 B
    float*          vals  = (float*)(ws + 239075328);           // 16,777,216 B
    unsigned short* wbuf  = (unsigned short*)(ws + 255852544);  // 8 MB bf16 weights
    float*          Gx    = Gf;                                  // reuse after encoder
    unsigned short* hc    = (unsigned short*)Gb;                 // reuse after encoder
    unsigned short* xemb  = xbuf;                                // reuse after keys/vals

    // --- convert weights fp32 -> bf16 (deterministic, every call) ---
    size_t wo = 0;
    auto cvtw = [&](int idx, size_t n) -> const unsigned short* {
        unsigned short* dst = wbuf + wo;
        wo += (n + 127) & ~(size_t)127;
        k_cvt_bf16<<<(unsigned)((n + 255) / 256), 256, 0, stream>>>(
            (const float*)d_in[idx], dst, (int)n);
        return dst;
    };
    const int Hs[3]  = {64, 128, 256};
    const int Ins[3] = {160, 256, 512};
    const int Ts[3]  = {1024, 512, 256};
    const unsigned short* WihE[3][2];
    const unsigned short* WhhE[3][2];
    for (int li = 0; li < 3; ++li)
        for (int d = 0; d < 2; ++d) {
            int base = 2 + li * 8 + d * 4;
            WihE[li][d] = cvtw(base,     (size_t)4 * Hs[li] * Ins[li]);
            WhhE[li][d] = cvtw(base + 1, (size_t)4 * Hs[li] * Hs[li]);
        }
    const unsigned short* WihD = cvtw(27, 1024 * 512);
    const unsigned short* WhhD = cvtw(28, 1024 * 256);
    const unsigned short* WqD  = cvtw(31, 256 * 256);
    const unsigned short* WkD  = cvtw(33, 256 * 512);
    const unsigned short* WvD  = cvtw(35, 256 * 512);
    const unsigned short* WoD  = cvtw(37, 1000 * 512);

    // --- encoder: 3 pBLSTM layers ---
    float* ybufs[2] = {ybufA, ybufB};
    for (int li = 0; li < 3; ++li) {
        int H = Hs[li], In = Ins[li], T = Ts[li], F = In / 2;
        const float* src;
        long sb, st;
        if (li == 0) { src = spectras; sb = 2048l * 80; st = 80; }
        else         { src = ybufs[(li + 1) & 1]; sb = F; st = 64l * F; }
        int total = T * 64 * In;
        k_fold_bf16<<<(total + 255) / 256, 256, 0, stream>>>(src, xbuf, F, sb, st, total);
        int N = 4 * H;
        int tiles = (T * 64 / 64) * (N / 16);
        const float* bihF = (const float*)d_in[2 + li * 8 + 2];
        const float* bhhF = (const float*)d_in[2 + li * 8 + 3];
        const float* bihB = (const float*)d_in[2 + li * 8 + 6];
        const float* bhhB = (const float*)d_in[2 + li * 8 + 7];
        k_gemm<false><<<(tiles + 7) / 8, 256, 0, stream>>>(xbuf, In, WihE[li][0], In,
            bihF, bhhF, Gf, N, T * 64, N, N, In);
        k_gemm<false><<<(tiles + 7) / 8, 256, 0, stream>>>(xbuf, In, WihE[li][1], In,
            bihB, bhhB, Gb, N, T * 64, N, N, In);
        if (H == 64)
            k_lstm<64><<<8, 512, 0, stream>>>(Gf, Gb, WhhE[li][0], WhhE[li][1],
                                              ybufs[li & 1], T);
        else if (H == 128)
            k_lstm<128><<<8, 512, 0, stream>>>(Gf, Gb, WhhE[li][0], WhhE[li][1],
                                               ybufs[li & 1], T);
        else
            k_lstm<256><<<8, 512, 0, stream>>>(Gf, Gb, WhhE[li][0], WhhE[li][1],
                                               ybufs[li & 1], T);
    }
    // features = ybufA : [256][64][512] fp32

    // --- keys / values projections (big GEMMs) ---
    k_cvt_bf16<<<(256 * 64 * 512 + 255) / 256, 256, 0, stream>>>(ybufA, xbuf,
                                                                 256 * 64 * 512);
    {
        int tiles = (16384 / 64) * (256 / 16);
        k_gemm<false><<<(tiles + 7) / 8, 256, 0, stream>>>(xbuf, 512, WkD, 512,
            (const float*)d_in[34], nullptr, keys, 256, 16384, 256, 256, 512);
        k_gemm<false><<<(tiles + 7) / 8, 256, 0, stream>>>(xbuf, 512, WvD, 512,
            (const float*)d_in[36], nullptr, vals, 256, 16384, 256, 256, 512);
    }

    // --- decoder input gates from embeddings (step-invariant part) ---
    k_gather_emb<<<(128 * 64 * 256 + 255) / 256, 256, 0, stream>>>(
        (const float*)d_in[26], seqs, xemb);
    {
        int tiles = (8192 / 64) * (1024 / 16);
        k_gemm<false><<<(tiles + 7) / 8, 256, 0, stream>>>(xemb, 256, WihD, 512,
            (const float*)d_in[29], (const float*)d_in[30], Gx, 1024,
            8192, 1024, 1024, 256);
    }

    // --- persistent attention decoder recurrence ---
    k_dec<<<4, 512, 0, stream>>>(Gx, keys, vals, WihD, WhhD, WqD,
                                 (const float*)d_in[32], hc);

    // --- final logits GEMM: [64*128, 512] x [1000, 512]^T -> d_out [B][S][V] ---
    {
        int tiles = (8192 / 64) * (1008 / 16);
        k_gemm<true><<<(tiles + 7) / 8, 256, 0, stream>>>(hc, 512, WoD, 512,
            (const float*)d_in[38], nullptr, (float*)d_out, 1000,
            8192, 1000, 1008, 512);
    }
}
